// TransformerBlock_Drop_63170378989985
// MI455X (gfx1250) — matrix operand order repaired and checked by reading, not compiled, not run on hardware
//
#include <hip/hip_runtime.h>
#include <hip/hip_bf16.h>

// ---------------------------------------------------------------------------
// Transformer block forward for MI455X (gfx1250).
//   bf16 WMMA (v_wmma_f32_16x16x32_bf16) + f32 accumulate everywhere.
//   Weights pre-transposed+converted to bf16 once; activations carried in
//   bf16 shadows so GEMM/attention staging is raw 16B copies, which go
//   through GLOBAL_LOAD_ASYNC_TO_LDS_B128 (ASYNCcnt) when available.
// ---------------------------------------------------------------------------

#define BATCH    2
#define SEQ      2048
#define DMODEL   1024
#define NHEADS   16
#define HEADDIM  64
#define FFDIM    4096

typedef __attribute__((ext_vector_type(16))) __bf16 v16bf;
typedef __attribute__((ext_vector_type(2)))  __bf16 v2bf;
typedef __attribute__((ext_vector_type(8)))  float  v8f;
typedef int v4i __attribute__((vector_size(16)));   // matches builtin param type

union BF16x16 { v16bf v; unsigned short u[16]; unsigned int q[8]; };

// ---- f32 -> bf16 (hardware cvt_pk when available) -------------------------
static __device__ inline unsigned int pack2bf(float lo, float hi) {
#if __has_builtin(__builtin_amdgcn_cvt_pk_bf16_f32)
  return __builtin_bit_cast(unsigned int, __builtin_amdgcn_cvt_pk_bf16_f32(lo, hi));
#else
  unsigned int xl = __float_as_uint(lo); xl += 0x7FFFu + ((xl >> 16) & 1u);
  unsigned int xh = __float_as_uint(hi); xh += 0x7FFFu + ((xh >> 16) & 1u);
  return (xl >> 16) | (xh & 0xFFFF0000u);
#endif
}
static __device__ inline unsigned short f2bf(float f) {
  return (unsigned short)pack2bf(f, 0.f);
}

// ---- async global->LDS 16B copy (CDNA5 ASYNCcnt path) ---------------------
#if __has_builtin(__builtin_amdgcn_global_load_async_to_lds_b128) && \
    __has_builtin(__builtin_amdgcn_s_wait_asynccnt)
#define USE_ASYNC_LDS 1
#else
#define USE_ASYNC_LDS 0
#endif

static __device__ inline void async_copy16(const unsigned short* g, unsigned short* l) {
#if USE_ASYNC_LDS
  __builtin_amdgcn_global_load_async_to_lds_b128(
      (__attribute__((address_space(1))) v4i*)g,
      (__attribute__((address_space(3))) v4i*)l, 0, 0);
#else
  *reinterpret_cast<uint4*>(l) = *reinterpret_cast<const uint4*>(g);
#endif
}
static __device__ inline void async_join() {
#if USE_ASYNC_LDS
  __builtin_amdgcn_s_wait_asynccnt(0);
#endif
}

static __device__ inline v8f wmma_bf16(const BF16x16& a, const BF16x16& b, v8f c) {
  // (neg_a, A, neg_b, B, c_mod, C, reuse_a, reuse_b)
  return __builtin_amdgcn_wmma_f32_16x16x32_bf16(false, a.v, false, b.v,
                                                 (short)0, c, false, false);
}

// ---------------------------------------------------------------------------
// Elementwise f32 -> bf16 (float4 in, uint2 out). n must be multiple of 1024.
// ---------------------------------------------------------------------------
__global__ void __launch_bounds__(256)
convert_bf16(const float* __restrict__ in, unsigned short* __restrict__ out) {
  const size_t i4 = (size_t)blockIdx.x * 256 + threadIdx.x;
  const float4 f = reinterpret_cast<const float4*>(in)[i4];
  reinterpret_cast<uint2*>(out)[i4] =
      make_uint2(pack2bf(f.x, f.y), pack2bf(f.z, f.w));
}

// ---------------------------------------------------------------------------
// W[K][N] f32 -> Wt[N][K] bf16 (32x32 LDS tile transpose). Block 256 thr.
// ---------------------------------------------------------------------------
__global__ void __launch_bounds__(256)
transpose_convert_bf16(const float* __restrict__ W, unsigned short* __restrict__ Wt,
                       int K, int N) {
  __shared__ float tile[32][33];
  const int n0 = blockIdx.x * 32, k0 = blockIdx.y * 32;
  const int tx = threadIdx.x & 31, ty = threadIdx.x >> 5;   // 32 x 8
  #pragma unroll
  for (int i = 0; i < 4; i++)
    tile[ty + i * 8][tx] = W[(size_t)(k0 + ty + i * 8) * N + n0 + tx];
  __syncthreads();
  #pragma unroll
  for (int i = 0; i < 4; i++)
    Wt[(size_t)(n0 + ty + i * 8) * K + k0 + tx] = f2bf(tile[tx][ty + i * 8]);
}

// ---------------------------------------------------------------------------
// GEMM: C[M,N] = act(A[M,K] @ Wt[N,K]^T + bias[N]),  A/Wt in bf16.
// Block = 256 thr = 8 waves; block tile 128x32; K-step 64; 4 WMMAs per sync.
// Staging = raw 16B async copies (no conversion, no transpose).
// Optional f32 and/or bf16 outputs.
// WMMA A/B operands use contiguous k order: element s of lane half h is k = 16h + s.
// ---------------------------------------------------------------------------
__global__ void __launch_bounds__(256)
gemm_bf16_wmma(const unsigned short* __restrict__ A,   // [M,K] bf16
               const unsigned short* __restrict__ Wt,  // [N,K] bf16 (pre-T)
               const float* __restrict__ bias,
               float* __restrict__ Cf,                 // optional f32 out
               unsigned short* __restrict__ Cb,        // optional bf16 out
               int M, int N, int K, int relu)
{
  __shared__ unsigned short Alds[128][64];  // 16 KB
  __shared__ unsigned short Blds[32][64];   //  4 KB

  const int t    = threadIdx.x;
  const int lane = t & 31;
  const int w    = t >> 5;
  const int ln   = lane & 15;
  const int lh   = lane >> 4;

  const int n0 = blockIdx.x * 32;
  const int m0 = blockIdx.y * 128;

  v8f acc0 = {}, acc1 = {};

  for (int kb = 0; kb < K; kb += 64) {
    // A slab: 128 rows x 128 B  -> 1024 x 16B chunks / 256 thr
    #pragma unroll
    for (int i = 0; i < 4; i++) {
      const int c = t + i * 256;
      const int r = c >> 3, off = (c & 7) * 8;          // off in bf16 elements
      async_copy16(&A[(size_t)(m0 + r) * K + kb + off], &Alds[r][off]);
    }
    // B slab: 32 rows x 128 B -> 256 x 16B chunks
    {
      const int r = t >> 3, off = (t & 7) * 8;
      async_copy16(&Wt[(size_t)(n0 + r) * K + kb + off], &Blds[r][off]);
    }
    async_join();
    __syncthreads();

    const int am = w * 16 + ln;
    #pragma unroll
    for (int kk = 0; kk < 2; kk++) {
      BF16x16 a, b0, b1;
      const unsigned int* ap0 =
          reinterpret_cast<const unsigned int*>(&Alds[am][kk * 32 + 16 * lh]);
      const unsigned int* ap1 =
          reinterpret_cast<const unsigned int*>(&Alds[am][kk * 32 + 16 * lh + 8]);
      #pragma unroll
      for (int j = 0; j < 4; j++) { a.q[j] = ap0[j]; a.q[4 + j] = ap1[j]; }
      const unsigned int* bp0 =
          reinterpret_cast<const unsigned int*>(&Blds[ln][kk * 32 + 16 * lh]);
      const unsigned int* bp1 =
          reinterpret_cast<const unsigned int*>(&Blds[16 + ln][kk * 32 + 16 * lh]);
      #pragma unroll
      for (int j = 0; j < 8; j++) { b0.q[j] = bp0[j]; b1.q[j] = bp1[j]; }
      acc0 = wmma_bf16(a, b0, acc0);
      acc1 = wmma_bf16(a, b1, acc1);
    }
    __syncthreads();
  }

  const float bv0 = bias[n0 + ln];
  const float bv1 = bias[n0 + 16 + ln];
  #pragma unroll
  for (int r = 0; r < 8; r++) {
    const int m = m0 + w * 16 + r + 8 * lh;
    float v0 = acc0[r] + bv0;
    float v1 = acc1[r] + bv1;
    if (relu) { v0 = fmaxf(v0, 0.f); v1 = fmaxf(v1, 0.f); }
    if (Cf) {
      Cf[(size_t)m * N + n0 + ln]      = v0;
      Cf[(size_t)m * N + n0 + 16 + ln] = v1;
    }
    if (Cb) {
      Cb[(size_t)m * N + n0 + ln]      = f2bf(v0);
      Cb[(size_t)m * N + n0 + 16 + ln] = f2bf(v1);
    }
  }
}

// ---------------------------------------------------------------------------
// Flash attention, one block per (b, head, 16-query tile). Q/K/V are bf16.
// 8 waves; wave w owns key chunk [it*256+32w, +32). V slab (256x64 bf16)
// staged via async copies; unions with the post-loop accW buffer.
// 1/sqrt(Dh) applied to f32 scores after the WMMA.
// ---------------------------------------------------------------------------
union AttnShared {
  float          accW[8][16][64];     // 32 KB, after the key loop
  unsigned short Vlds[256][64];       // 32 KB, inside the key loop
};

__global__ void __launch_bounds__(256)
attention_wmma(const unsigned short* __restrict__ Qp,
               const unsigned short* __restrict__ Kp,
               const unsigned short* __restrict__ Vp,
               unsigned short* __restrict__ O)       // bf16 out [B,S,D]
{
  const int qt = blockIdx.x;
  const int hh = blockIdx.y;
  const int bb = blockIdx.z;

  const int t    = threadIdx.x;
  const int lane = t & 31;
  const int w    = t >> 5;
  const int ln   = lane & 15;
  const int lh   = lane >> 4;

  __shared__ AttnShared sh;                        // 32 KB (union)
  __shared__ unsigned short Plds[8][16][32];       // 8 KB
  __shared__ float mW[8][16], lW[8][16];
  __shared__ float Mrow[16], Lrow[16];

  const size_t base = ((size_t)bb * SEQ) * DMODEL + (size_t)hh * HEADDIM;

  // Q tile A-operands (bf16 direct loads, no conversion)
  BF16x16 aq[2];
  {
    const unsigned short* qp = Qp + base + (size_t)(qt * 16 + ln) * DMODEL;
    #pragma unroll
    for (int c = 0; c < 2; c++) {
      const uint4 f0 = *reinterpret_cast<const uint4*>(&qp[c * 32 + 16 * lh]);
      const uint4 f1 = *reinterpret_cast<const uint4*>(&qp[c * 32 + 16 * lh + 8]);
      aq[c].q[0] = f0.x; aq[c].q[1] = f0.y; aq[c].q[2] = f0.z; aq[c].q[3] = f0.w;
      aq[c].q[4] = f1.x; aq[c].q[5] = f1.y; aq[c].q[6] = f1.z; aq[c].q[7] = f1.w;
    }
  }

  float mrun[8], lrun[8];
  v8f acc[4];
  #pragma unroll
  for (int r = 0; r < 8; r++) { mrun[r] = -1e30f; lrun[r] = 0.f; }
  #pragma unroll
  for (int d = 0; d < 4; d++) acc[d] = (v8f){};

  for (int it = 0; it < SEQ / 256; ++it) {        // 8 iterations
    const int kcb = it * 256;
    const int kc  = kcb + w * 32;

    // ---- stage V slab (256 keys x 64 d bf16 = 32 KB) via async copies ----
    #pragma unroll
    for (int i = 0; i < 8; i++) {                 // 2048 x 16B / 256 thr
      const int c  = t + i * 256;
      const int kk = c >> 3, off = (c & 7) * 8;
      async_copy16(&Vp[base + (size_t)(kcb + kk) * DMODEL + off],
                   &sh.Vlds[kk][off]);
    }

    // ---- scores: S(16x32) = Q @ K^T, two 16x16 tiles ----
    v8f s[2] = { (v8f){}, (v8f){} };
    #pragma unroll
    for (int kt = 0; kt < 2; kt++) {
      const unsigned short* kp =
          Kp + base + (size_t)(kc + kt * 16 + ln) * DMODEL;
      #pragma unroll
      for (int c = 0; c < 2; c++) {
        BF16x16 bk;
        const uint4 f0 = *reinterpret_cast<const uint4*>(&kp[c * 32 + 16 * lh]);
        const uint4 f1 = *reinterpret_cast<const uint4*>(&kp[c * 32 + 16 * lh + 8]);
        bk.q[0] = f0.x; bk.q[1] = f0.y; bk.q[2] = f0.z; bk.q[3] = f0.w;
        bk.q[4] = f1.x; bk.q[5] = f1.y; bk.q[6] = f1.z; bk.q[7] = f1.w;
        s[kt] = wmma_bf16(aq[c], bk, s[kt]);
      }
    }

    // ---- online softmax (row m = r + 8*lh); 1/8 scale applied here ----
    float corr[8];
    #pragma unroll
    for (int r = 0; r < 8; r++) {
      const float s0 = s[0][r] * 0.125f;
      const float s1 = s[1][r] * 0.125f;
      float tmax = fmaxf(s0, s1);
      tmax = fmaxf(tmax, __shfl_xor(tmax, 1, 16));
      tmax = fmaxf(tmax, __shfl_xor(tmax, 2, 16));
      tmax = fmaxf(tmax, __shfl_xor(tmax, 4, 16));
      tmax = fmaxf(tmax, __shfl_xor(tmax, 8, 16));
      const float mnew = fmaxf(mrun[r], tmax);
      corr[r] = __expf(mrun[r] - mnew);
      const float p0 = __expf(s0 - mnew);
      const float p1 = __expf(s1 - mnew);
      float psum = p0 + p1;
      psum += __shfl_xor(psum, 1, 16);
      psum += __shfl_xor(psum, 2, 16);
      psum += __shfl_xor(psum, 4, 16);
      psum += __shfl_xor(psum, 8, 16);
      lrun[r] = lrun[r] * corr[r] + psum;
      mrun[r] = mnew;
      Plds[w][r + 8 * lh][ln]      = f2bf(p0);
      Plds[w][r + 8 * lh][16 + ln] = f2bf(p1);
    }
    #pragma unroll
    for (int d = 0; d < 4; d++)
      #pragma unroll
      for (int r = 0; r < 8; r++) acc[d][r] *= corr[r];
    async_join();
    __syncthreads();   // publishes Vlds + Plds

    // ---- PV: acc(16x64) += P(16x32) @ V(32x64) ----
    BF16x16 ap;
    {
      const unsigned int* p0 =
          reinterpret_cast<const unsigned int*>(&Plds[w][ln][16 * lh]);
      const unsigned int* p1 =
          reinterpret_cast<const unsigned int*>(&Plds[w][ln][16 * lh + 8]);
      #pragma unroll
      for (int j = 0; j < 4; j++) { ap.q[j] = p0[j]; ap.q[4 + j] = p1[j]; }
    }
    #pragma unroll
    for (int dt = 0; dt < 4; dt++) {
      BF16x16 bv;
      #pragma unroll
      for (int i = 0; i < 16; i++)
        bv.u[i] = sh.Vlds[w * 32 + 16 * lh + i][dt * 16 + ln];
      acc[dt] = wmma_bf16(ap, bv, acc[dt]);
    }
    __syncthreads();
  }

  // ---- cross-wave merge (accW reuses the Vlds LDS space) ----
  #pragma unroll
  for (int r = 0; r < 8; r++) {
    const int m = r + 8 * lh;
    if (ln == 0) { mW[w][m] = mrun[r]; lW[w][m] = lrun[r]; }
    #pragma unroll
    for (int dt = 0; dt < 4; dt++)
      sh.accW[w][m][dt * 16 + ln] = acc[dt][r];
  }
  __syncthreads();

  if (t < 16) {
    float M = -1e30f;
    #pragma unroll
    for (int ww = 0; ww < 8; ww++) M = fmaxf(M, mW[ww][t]);
    float L = 0.f;
    #pragma unroll
    for (int ww = 0; ww < 8; ww++) L += lW[ww][t] * __expf(mW[ww][t] - M);
    Mrow[t] = M; Lrow[t] = L;
  }
  __syncthreads();

  {
    const int m = t >> 4, d4 = t & 15;           // 256 x (4 bf16) outputs
    float sx = 0.f, sy = 0.f, sz = 0.f, sw = 0.f;
    #pragma unroll
    for (int ww = 0; ww < 8; ww++) {
      const float e = __expf(mW[ww][m] - Mrow[m]);
      sx += sh.accW[ww][m][d4 * 4 + 0] * e;
      sy += sh.accW[ww][m][d4 * 4 + 1] * e;
      sz += sh.accW[ww][m][d4 * 4 + 2] * e;
      sw += sh.accW[ww][m][d4 * 4 + 3] * e;
    }
    const float inv = 1.f / Lrow[m];
    *reinterpret_cast<uint2*>(
        &O[base + (size_t)(qt * 16 + m) * DMODEL + d4 * 4]) =
        make_uint2(pack2bf(sx * inv, sy * inv), pack2bf(sz * inv, sw * inv));
  }
}

// ---------------------------------------------------------------------------
// out = LayerNorm(a + b) * g + beta; optional bf16 shadow. One float4/thread.
// ---------------------------------------------------------------------------
__global__ void __launch_bounds__(256)
add_layernorm(const float* __restrict__ a, const float* __restrict__ b,
              const float* __restrict__ g, const float* __restrict__ be,
              float* __restrict__ out, unsigned short* __restrict__ outb)
{
  __shared__ float red[256];
  const int row = blockIdx.x;
  const int t   = threadIdx.x;

  const float4 xa = reinterpret_cast<const float4*>(a + (size_t)row * DMODEL)[t];
  const float4 xb = reinterpret_cast<const float4*>(b + (size_t)row * DMODEL)[t];
  float4 x;
  x.x = xa.x + xb.x; x.y = xa.y + xb.y; x.z = xa.z + xb.z; x.w = xa.w + xb.w;

  red[t] = x.x + x.y + x.z + x.w;
  __syncthreads();
  for (int off = 128; off; off >>= 1) {
    if (t < off) red[t] += red[t + off];
    __syncthreads();
  }
  const float mu = red[0] * (1.f / DMODEL);
  __syncthreads();

  const float dx = x.x - mu, dy = x.y - mu, dz = x.z - mu, dw = x.w - mu;
  red[t] = dx * dx + dy * dy + dz * dz + dw * dw;
  __syncthreads();
  for (int off = 128; off; off >>= 1) {
    if (t < off) red[t] += red[t + off];
    __syncthreads();
  }
  const float inv = rsqrtf(red[0] * (1.f / DMODEL) + 1e-5f);

  const float4 gv = reinterpret_cast<const float4*>(g)[t];
  const float4 bv = reinterpret_cast<const float4*>(be)[t];
  float4 o;
  o.x = dx * inv * gv.x + bv.x;
  o.y = dy * inv * gv.y + bv.y;
  o.z = dz * inv * gv.z + bv.z;
  o.w = dw * inv * gv.w + bv.w;
  reinterpret_cast<float4*>(out + (size_t)row * DMODEL)[t] = o;
  if (outb)
    reinterpret_cast<uint2*>(outb + (size_t)row * DMODEL)[t] =
        make_uint2(pack2bf(o.x, o.y), pack2bf(o.z, o.w));
}

// ---------------------------------------------------------------------------
extern "C" void kernel_launch(void* const* d_in, const int* in_sizes, int n_in,
                              void* d_out, int out_size, void* d_ws, size_t ws_size,
                              hipStream_t stream) {
  (void)in_sizes; (void)n_in; (void)out_size; (void)ws_size;
  const float* value = (const float*)d_in[0];
  const float* key   = (const float*)d_in[1];
  const float* query = (const float*)d_in[2];
  const float* wq = (const float*)d_in[3];  const float* bq = (const float*)d_in[4];
  const float* wk = (const float*)d_in[5];  const float* bk = (const float*)d_in[6];
  const float* wv = (const float*)d_in[7];  const float* bv = (const float*)d_in[8];
  const float* wo = (const float*)d_in[9];  const float* bo = (const float*)d_in[10];
  const float* g1 = (const float*)d_in[11]; const float* beta1 = (const float*)d_in[12];
  const float* w1 = (const float*)d_in[13]; const float* b1 = (const float*)d_in[14];
  const float* w2 = (const float*)d_in[15]; const float* b2 = (const float*)d_in[16];
  const float* g2 = (const float*)d_in[17]; const float* beta2 = (const float*)d_in[18];

  const int M = BATCH * SEQ;                   // 4096 rows
  const size_t SL = (size_t)M * DMODEL;        // 4M elements

  // f32 region: 3 x 16 MB
  float* F_oproj = (float*)d_ws;
  float* F_x1    = F_oproj + SL;
  float* F_ff2   = F_oproj + 2 * SL;
  // bf16 region
  unsigned short* hb   = (unsigned short*)(F_oproj + 3 * SL);
  unsigned short* BQin = hb;                 // converted inputs [M,D]
  unsigned short* BKin = hb + 1 * SL;
  unsigned short* BVin = hb + 2 * SL;
  unsigned short* PQ   = hb + 3 * SL;        // projections [M,D]
  unsigned short* PK   = hb + 4 * SL;
  unsigned short* PV   = hb + 5 * SL;
  unsigned short* ATT  = hb + 6 * SL;        // attention out [M,D]
  unsigned short* X1B  = hb + 7 * SL;        // LN1 out shadow [M,D]
  unsigned short* WqT  = hb + 8 * SL;        // [N,K] transposed weights
  unsigned short* WkT  = WqT + (size_t)DMODEL * DMODEL;
  unsigned short* WvT  = WkT + (size_t)DMODEL * DMODEL;
  unsigned short* WoT  = WvT + (size_t)DMODEL * DMODEL;
  unsigned short* W1T  = WoT + (size_t)DMODEL * DMODEL;   // [FF,D]
  unsigned short* W2T  = W1T + (size_t)FFDIM * DMODEL;    // [D,FF]
  unsigned short* FF1B = W2T + (size_t)DMODEL * FFDIM;    // [M,FF]

  const dim3 blk(256);
  const dim3 gP(DMODEL / 32, M / 128);
  const dim3 gF1(FFDIM / 32, M / 128);
  const dim3 gAtt(SEQ / 16, NHEADS, BATCH);

  // one-time conversions
  convert_bf16<<<(SL / 4 + 255) / 256, blk, 0, stream>>>(query, BQin);
  convert_bf16<<<(SL / 4 + 255) / 256, blk, 0, stream>>>(key,   BKin);
  convert_bf16<<<(SL / 4 + 255) / 256, blk, 0, stream>>>(value, BVin);
  transpose_convert_bf16<<<dim3(DMODEL/32, DMODEL/32), blk, 0, stream>>>(wq, WqT, DMODEL, DMODEL);
  transpose_convert_bf16<<<dim3(DMODEL/32, DMODEL/32), blk, 0, stream>>>(wk, WkT, DMODEL, DMODEL);
  transpose_convert_bf16<<<dim3(DMODEL/32, DMODEL/32), blk, 0, stream>>>(wv, WvT, DMODEL, DMODEL);
  transpose_convert_bf16<<<dim3(DMODEL/32, DMODEL/32), blk, 0, stream>>>(wo, WoT, DMODEL, DMODEL);
  transpose_convert_bf16<<<dim3(FFDIM/32,  DMODEL/32), blk, 0, stream>>>(w1, W1T, DMODEL, FFDIM);
  transpose_convert_bf16<<<dim3(DMODEL/32, FFDIM/32),  blk, 0, stream>>>(w2, W2T, FFDIM, DMODEL);

  // QKV projections (bf16 out only)
  gemm_bf16_wmma<<<gP, blk, 0, stream>>>(BQin, WqT, bq, nullptr, PQ, M, DMODEL, DMODEL, 0);
  gemm_bf16_wmma<<<gP, blk, 0, stream>>>(BKin, WkT, bk, nullptr, PK, M, DMODEL, DMODEL, 0);
  gemm_bf16_wmma<<<gP, blk, 0, stream>>>(BVin, WvT, bv, nullptr, PV, M, DMODEL, DMODEL, 0);
  // attention (bf16 in/out)
  attention_wmma<<<gAtt, blk, 0, stream>>>(PQ, PK, PV, ATT);
  // output projection (f32 out for residual+LN)
  gemm_bf16_wmma<<<gP, blk, 0, stream>>>(ATT, WoT, bo, F_oproj, nullptr, M, DMODEL, DMODEL, 0);
  // x1 = LN(oproj + query): f32 + bf16 shadow
  add_layernorm<<<M, blk, 0, stream>>>(F_oproj, query, g1, beta1, F_x1, X1B);
  // ff1 = relu(x1 @ w1 + b1): bf16 only
  gemm_bf16_wmma<<<gF1, blk, 0, stream>>>(X1B, W1T, b1, nullptr, FF1B, M, FFDIM, DMODEL, 1);
  // ff2 = ff1 @ w2 + b2: f32
  gemm_bf16_wmma<<<gP, blk, 0, stream>>>(FF1B, W2T, b2, F_ff2, nullptr, M, DMODEL, FFDIM, 0);
  // out = LN(ff2 + x1)
  add_layernorm<<<M, blk, 0, stream>>>(F_ff2, F_x1, g2, beta2, (float*)d_out, nullptr);
}
